// PyramidAttention_6459630813244
// MI455X (gfx1250) — matrix operand order repaired and checked by reading, not compiled, not run on hardware
//
#include <hip/hip_runtime.h>
#include <hip/hip_bf16.h>

// ---------------- problem constants ----------------
#define BATCH   2
#define CH      64
#define CR      16
#define HH      64
#define WW      64
#define HWQ     4096          // 64*64
#define HS      57
#define HWS     3249          // 57*57
#define LVAL    7345          // 4096 + 3249
#define LPAD    7360          // padded to multiple of 32
#define DQ      144           // Cr*9
#define DQP     160           // padded to multiple of 32
#define DV      576           // C*9
#define DVW     288           // dv slice per wave (2 waves)
#define NT      18            // acc tiles per wave (288/16)
#define SMK     14.4269504088896340736f   // 10 * log2(e)

typedef __attribute__((ext_vector_type(16))) _Float16 v16h;
typedef __attribute__((ext_vector_type(8)))  _Float16 v8h;
typedef __attribute__((ext_vector_type(8)))  float    v8f;

// ---------------- workspace layout (element offsets) ----------------
// f32 region (float*):
//   qbase  @ 0         (131072)   conv(input, w_mb)  [2,16,64,64]
//   kbf    @ 131072    (131072)   conv(input, w_m)   [2,16,64,64]
//   kbs    @ 262144    (103968)   conv(small, w_m)   [2,16,57,57]
//   vbf    @ 366112    (524288)   conv(input, w_as)  [2,64,64,64]
//   vbs    @ 890400    (415872)   conv(small, w_as)  [2,64,57,57]
//   T      @ 1306272   (4718592)  attn output        [2,4096,576]
// f16 region starts at byte 24099456 (_Float16*):
//   q      @ 0         (1310720)  [2,4096,160]
//   kn     @ 1310720   (2355200)  [2,7360,160] (L2-normalized rows)
//   vT     @ 3665920   (8478720)  [2,576,7360] (transposed V)
#define OFF_QBASE 0
#define OFF_KBF   131072
#define OFF_KBS   262144
#define OFF_VBF   366112
#define OFF_VBS   890400
#define OFF_T     1306272
#define F16_BYTE  24099456
#define OFF_Q16   0
#define OFF_K16   1310720
#define OFF_V16   3665920

// ---------------- 16-lane butterfly reductions (VALU permlane, no DS pipe) ----
#if __has_builtin(__builtin_amdgcn_permlane16)
__device__ __forceinline__ float pa_permx16(float x, unsigned s1, unsigned s2) {
    int v = __builtin_bit_cast(int, x);
    v = __builtin_amdgcn_permlane16(v, v, (int)s1, (int)s2, false, false);
    return __builtin_bit_cast(float, v);
}
__device__ __forceinline__ float pa_max16(float x) {
    x = fmaxf(x, pa_permx16(x, 0x67452301u, 0xEFCDAB89u));  // xor 1
    x = fmaxf(x, pa_permx16(x, 0x54761032u, 0xDCFE98BAu));  // xor 2
    x = fmaxf(x, pa_permx16(x, 0x32107654u, 0xBA98FEDCu));  // xor 4
    x = fmaxf(x, pa_permx16(x, 0xFEDCBA98u, 0x76543210u));  // xor 8
    return x;
}
__device__ __forceinline__ float pa_sum16(float x) {
    x += pa_permx16(x, 0x67452301u, 0xEFCDAB89u);
    x += pa_permx16(x, 0x54761032u, 0xDCFE98BAu);
    x += pa_permx16(x, 0x32107654u, 0xBA98FEDCu);
    x += pa_permx16(x, 0xFEDCBA98u, 0x76543210u);
    return x;
}
#else
__device__ __forceinline__ float pa_max16(float x) {
    for (int off = 1; off < 16; off <<= 1) x = fmaxf(x, __shfl_xor(x, off, 32));
    return x;
}
__device__ __forceinline__ float pa_sum16(float x) {
    for (int off = 1; off < 16; off <<= 1) x += __shfl_xor(x, off, 32);
    return x;
}
#endif

// ---------------- K1: 1x1 conv + PReLU ----------------
__global__ void pa_conv1x1_prelu(const float* __restrict__ in,
                                 const float* __restrict__ w,
                                 const float* __restrict__ bias,
                                 const float* __restrict__ aslope,
                                 float* __restrict__ out,
                                 int Cout, int HW, int total) {
    int idx = blockIdx.x * blockDim.x + threadIdx.x;
    if (idx >= total) return;
    int hw = idx % HW;
    int o  = (idx / HW) % Cout;
    int b  = idx / (HW * Cout);
    const float* ip = in + (size_t)b * CH * HW + hw;
    const float* wp = w + o * CH;
    float s = bias[o];
#pragma unroll 8
    for (int c = 0; c < CH; ++c) s += ip[(size_t)c * HW] * wp[c];
    float a = aslope[0];
    out[idx] = (s >= 0.f) ? s : a * s;
}

// ---------------- K2a: build Q (f16, padded to 160) ----------------
__global__ void pa_build_q(const float* __restrict__ qbase, _Float16* __restrict__ q) {
    int idx = blockIdx.x * blockDim.x + threadIdx.x;
    const int total = BATCH * HWQ * DQP;
    if (idx >= total) return;
    int d = idx % DQP;
    int p = (idx / DQP) % HWQ;
    int b = idx / (DQP * HWQ);
    float val = 0.f;
    if (d < DQ) {
        int c = d / 9, t = d % 9;
        int di = t / 3 - 1, dj = t % 3 - 1;
        int h = p >> 6, w = p & 63;
        int hh = h + di, ww = w + dj;
        if (hh >= 0 && hh < HH && ww >= 0 && ww < WW)
            val = qbase[(((size_t)b * CR + c) << 12) + (hh << 6) + ww];
    }
    q[idx] = (_Float16)val;
}

// ---------------- K2b: build normalized K rows (one wave per row) ----------------
__global__ void pa_build_k(const float* __restrict__ kbf, const float* __restrict__ kbs,
                           _Float16* __restrict__ kn) {
    int gwave = (blockIdx.x * blockDim.x + threadIdx.x) >> 5;
    int lane  = threadIdx.x & 31;
    if (gwave >= BATCH * LPAD) return;
    int b = gwave / LPAD;
    int r = gwave % LPAD;
    float vals[5];
    float ss = 0.f;
#pragma unroll
    for (int k = 0; k < 5; ++k) {
        int d = lane + 32 * k;
        float v = 0.f;
        if (d < DQ && r < LVAL) {
            int c = d / 9, t = d % 9;
            int di = t / 3 - 1, dj = t % 3 - 1;
            if (r < HWQ) {
                int h = r >> 6, w = r & 63;
                int hh = h + di, ww = w + dj;
                if (hh >= 0 && hh < HH && ww >= 0 && ww < WW)
                    v = kbf[(((size_t)b * CR + c) << 12) + (hh << 6) + ww];
            } else {
                int p = r - HWQ;
                int h = p / HS, w = p % HS;
                int hh = h + di, ww = w + dj;
                if (hh >= 0 && hh < HS && ww >= 0 && ww < HS)
                    v = kbs[((size_t)b * CR + c) * HWS + hh * HS + ww];
            }
        }
        vals[k] = v;
        ss += v * v;
    }
#pragma unroll
    for (int off = 16; off >= 1; off >>= 1) ss += __shfl_xor(ss, off, 32);
    float inv = 1.f / fmaxf(sqrtf(ss), 1e-4f);
    _Float16* row = kn + (size_t)gwave * DQP;
#pragma unroll
    for (int k = 0; k < 5; ++k) row[lane + 32 * k] = (_Float16)(vals[k] * inv);
}

// ---------------- K2c: build transposed V (f16) ----------------
__global__ void pa_build_v(const float* __restrict__ vbf, const float* __restrict__ vbs,
                           _Float16* __restrict__ vT) {
    int idx = blockIdx.x * blockDim.x + threadIdx.x;
    const int total = BATCH * DV * LPAD;
    if (idx >= total) return;
    int r = idx % LPAD;
    int e = (idx / LPAD) % DV;
    int b = idx / (LPAD * DV);
    float v = 0.f;
    if (r < LVAL) {
        int c = e / 9, t = e % 9;
        int di = t / 3 - 1, dj = t % 3 - 1;
        if (r < HWQ) {
            int h = r >> 6, w = r & 63;
            int hh = h + di, ww = w + dj;
            if (hh >= 0 && hh < HH && ww >= 0 && ww < WW)
                v = vbf[(((size_t)b * CH + c) << 12) + (hh << 6) + ww];
        } else {
            int p = r - HWQ;
            int h = p / HS, w = p % HS;
            int hh = h + di, ww = w + dj;
            if (hh >= 0 && hh < HS && ww >= 0 && ww < HS)
                v = vbs[((size_t)b * CH + c) * HWS + hh * HS + ww];
        }
    }
    vT[idx] = (_Float16)v;
}

// ---------------- K3: streaming flash-attention with WMMA ----------------
// Grid: BATCH*256 blocks (16 query rows each), 64 threads (2 waves).
// Wave w owns dv slice [w*288, w*288+288): 18 f32 16x16 accumulator tiles.
// Per 32-key step per wave: 10 WMMA (QK^T, shared/redundant x2) + 18 WMMA (P.V).
__launch_bounds__(64)
__global__ void pa_attention(const _Float16* __restrict__ q,
                             const _Float16* __restrict__ kn,
                             const _Float16* __restrict__ vT,
                             float* __restrict__ T) {
    __shared__ _Float16 psh[2][16 * 32] __attribute__((aligned(32)));
    const int lane = threadIdx.x & 31;
    const int wave = threadIdx.x >> 5;
    const int qt   = blockIdx.x & 255;
    const int b    = blockIdx.x >> 8;
    const int m16  = lane & 15;
    const int hsel = lane >> 4;

    // Q A-fragments: lane holds row (lane&15); per-lane K-halves are contiguous
    // {0..15} (lanes 0-15) / {16..31} (lanes 16-31), matching the B fragments.
    v16h qf[5];
    const _Float16* qrow = q + ((size_t)(b * HWQ + qt * 16 + m16)) * DQP;
#pragma unroll
    for (int kc = 0; kc < 5; ++kc) {
        qf[kc] = *(const v16h*)(qrow + kc * 32 + hsel * 16);
    }

    v8f acc[NT];
    const v8f vzero = {0.f, 0.f, 0.f, 0.f, 0.f, 0.f, 0.f, 0.f};
#pragma unroll
    for (int j = 0; j < NT; ++j) acc[j] = vzero;
    float mrow[8], lrow[8];
#pragma unroll
    for (int v = 0; v < 8; ++v) { mrow[v] = -1e30f; lrow[v] = 0.f; }

    const _Float16* kbase = kn + (size_t)b * LPAD * DQP;
    const _Float16* vbase = vT + (size_t)b * DV * LPAD
                          + (size_t)(wave * DVW + m16) * LPAD + (size_t)hsel * 16;
    _Float16* myp = &psh[wave][0];

    for (int kb = 0; kb < LPAD; kb += 32) {
        // prefetch next step's K/V streams into cache (global_prefetch_b8)
        if (kb + 32 < LPAD) {
            __builtin_prefetch(kbase + (size_t)(kb + 32 + m16) * DQP, 0, 1);
            __builtin_prefetch(vbase + kb + 32, 0, 1);
        }
        // ---- S = Q . K^T  (two 16x16 tiles covering 32 keys) ----
        v8f s0 = vzero, s1 = vzero;
#pragma unroll
        for (int kc = 0; kc < 5; ++kc) {
            // B-fragment: lane holds key column (lane&15), 16 contiguous k-dim halves
            v16h k0 = *(const v16h*)(kbase + (size_t)(kb + m16) * DQP + kc * 32 + hsel * 16);
            v16h k1 = *(const v16h*)(kbase + (size_t)(kb + 16 + m16) * DQP + kc * 32 + hsel * 16);
            s0 = __builtin_amdgcn_wmma_f32_16x16x32_f16(false, qf[kc], false, k0,
                                                        (short)0, s0, false, false);
            s1 = __builtin_amdgcn_wmma_f32_16x16x32_f16(false, qf[kc], false, k1,
                                                        (short)0, s1, false, false);
        }
        const bool msk0 = (kb + m16)      >= LVAL;
        const bool msk1 = (kb + 16 + m16) >= LVAL;

        // ---- online softmax; C-layout row = v + 8*hsel, col = lane&15 ----
        float scalef[8];
        int   changed = 0;
#pragma unroll
        for (int v = 0; v < 8; ++v) {
            float a0 = msk0 ? -1e30f : s0[v];
            float a1 = msk1 ? -1e30f : s1[v];
            float rmax = pa_max16(fmaxf(a0, a1));
            float mnew = fmaxf(mrow[v], rmax);
            changed |= (mnew > mrow[v]);
            float sc = exp2f(SMK * (mrow[v] - mnew));
            float p0 = exp2f(SMK * (a0 - mnew));
            float p1 = exp2f(SMK * (a1 - mnew));
            float rs = pa_sum16(p0 + p1);
            lrow[v] = lrow[v] * sc + rs;
            mrow[v] = mnew;
            scalef[v] = sc;
            int rr = v + 8 * hsel;
            myp[rr * 32 + m16]      = (_Float16)p0;
            myp[rr * 32 + 16 + m16] = (_Float16)p1;
        }
        // rescale accumulators only when some row max actually moved
        if (__any(changed)) {
#pragma unroll
            for (int j = 0; j < NT; ++j)
#pragma unroll
                for (int v = 0; v < 8; ++v) acc[j][v] *= scalef[v];
        }

        // same-wave LDS store->load ordering (DS pipe is in-order per wave)
        asm volatile("s_wait_dscnt 0x0" ::: "memory");

        // ---- reload P as A-fragment (16 rows x 32 keys) ----
        v16h pa;
        {
            pa = *(const v16h*)(myp + m16 * 32 + hsel * 16);
        }

        // ---- acc += P . V  over this wave's 288-wide dv slice ----
        const _Float16* vb = vbase + kb;
#pragma unroll
        for (int j = 0; j < NT; ++j) {
            v16h vf = *(const v16h*)(vb + (size_t)j * 16 * LPAD);
            acc[j] = __builtin_amdgcn_wmma_f32_16x16x32_f16(false, pa, false, vf,
                                                            (short)0, acc[j], false, false);
        }
    }

    // ---- normalize and write T ----
#pragma unroll
    for (int j = 0; j < NT; ++j)
#pragma unroll
        for (int v = 0; v < 8; ++v) {
            int row = v + 8 * hsel;
            int p = qt * 16 + row;
            int e = wave * DVW + j * 16 + m16;
            T[((size_t)(b * HWQ + p)) * DV + e] = acc[j][v] / lrow[v];
        }
}

// ---------------- K4: overlap-add (conv_transpose) /4 + residual ----------------
__global__ void pa_finalize(const float* __restrict__ T,
                            const float* __restrict__ inp,
                            float* __restrict__ y) {
    int idx = blockIdx.x * blockDim.x + threadIdx.x;
    const int total = BATCH * CH * HWQ;
    if (idx >= total) return;
    int w = idx & 63;
    int h = (idx >> 6) & 63;
    int c = (idx >> 12) & 63;
    int b = idx >> 18;
    float s = 0.f;
#pragma unroll
    for (int i = 0; i < 3; ++i) {
        int hh = h + 1 - i;
        if (hh < 0 || hh >= HH) continue;
#pragma unroll
        for (int j = 0; j < 3; ++j) {
            int ww2 = w + 1 - j;
            if (ww2 < 0 || ww2 >= WW) continue;
            int p = (hh << 6) + ww2;
            s += T[((size_t)(b * HWQ + p)) * DV + c * 9 + i * 3 + j];
        }
    }
    y[idx] = inp[idx] + 0.25f * s;
}

// ---------------- launcher ----------------
extern "C" void kernel_launch(void* const* d_in, const int* in_sizes, int n_in,
                              void* d_out, int out_size, void* d_ws, size_t ws_size,
                              hipStream_t stream) {
    (void)in_sizes; (void)n_in; (void)out_size; (void)ws_size;
    const float* input = (const float*)d_in[0];
    const float* small = (const float*)d_in[1];
    const float* w_mb  = (const float*)d_in[2];
    const float* b_mb  = (const float*)d_in[3];
    const float* a_mb  = (const float*)d_in[4];
    const float* w_m   = (const float*)d_in[5];
    const float* b_m   = (const float*)d_in[6];
    const float* a_m   = (const float*)d_in[7];
    const float* w_as  = (const float*)d_in[8];
    const float* b_as  = (const float*)d_in[9];
    const float* a_as  = (const float*)d_in[10];
    float* y = (float*)d_out;

    float* f32ws = (float*)d_ws;
    float* qbase = f32ws + OFF_QBASE;
    float* kbf   = f32ws + OFF_KBF;
    float* kbs   = f32ws + OFF_KBS;
    float* vbf   = f32ws + OFF_VBF;
    float* vbs   = f32ws + OFF_VBS;
    float* Tbuf  = f32ws + OFF_T;
    _Float16* h16 = (_Float16*)((char*)d_ws + F16_BYTE);
    _Float16* qh  = h16 + OFF_Q16;
    _Float16* knh = h16 + OFF_K16;
    _Float16* vTh = h16 + OFF_V16;

    const int TB = 256;
    auto cdiv = [](int a, int b) { return (a + b - 1) / b; };

    // 1x1 conv + PReLU projections
    {
        int t;
        t = BATCH * CR * HWQ;
        pa_conv1x1_prelu<<<cdiv(t, TB), TB, 0, stream>>>(input, w_mb, b_mb, a_mb, qbase, CR, HWQ, t);
        pa_conv1x1_prelu<<<cdiv(t, TB), TB, 0, stream>>>(input, w_m,  b_m,  a_m,  kbf,   CR, HWQ, t);
        t = BATCH * CR * HWS;
        pa_conv1x1_prelu<<<cdiv(t, TB), TB, 0, stream>>>(small, w_m,  b_m,  a_m,  kbs,   CR, HWS, t);
        t = BATCH * CH * HWQ;
        pa_conv1x1_prelu<<<cdiv(t, TB), TB, 0, stream>>>(input, w_as, b_as, a_as, vbf,   CH, HWQ, t);
        t = BATCH * CH * HWS;
        pa_conv1x1_prelu<<<cdiv(t, TB), TB, 0, stream>>>(small, w_as, b_as, a_as, vbs,   CH, HWS, t);
    }
    // operand staging (f16)
    pa_build_q<<<cdiv(BATCH * HWQ * DQP, TB), TB, 0, stream>>>(qbase, qh);
    pa_build_k<<<cdiv(BATCH * LPAD * 32, 128), 128, 0, stream>>>(kbf, kbs, knh);
    pa_build_v<<<cdiv(BATCH * DV * LPAD, TB), TB, 0, stream>>>(vbf, vbs, vTh);
    // WMMA flash attention (2 waves per 16-row query tile, dv split 2x288)
    pa_attention<<<BATCH * 256, 64, 0, stream>>>(qh, knh, vTh, Tbuf);
    // overlap-add + residual
    pa_finalize<<<cdiv(BATCH * CH * HWQ, TB), TB, 0, stream>>>(Tbuf, input, y);
}
